// LocallyLinearNN_83657372991736
// MI455X (gfx1250) — matrix operand order repaired and checked by reading, not compiled, not run on hardware
//
#include <hip/hip_runtime.h>
#include <hip/hip_bf16.h>

// ---------------- constants (from reference) ----------------
#define BATCH   4096
#define TSTEPS  64
#define NDIM    128          // 2 * N_POINTS
#define NCTL    2
#define HID     1024
#define KP1     160          // 130 padded up to 5 * 32 for bf16 WMMA K-steps
#define NB      256          // ELEMS_B columns of W3 that are actually used
#define GENSTR  (65 * NDIM)  // 8320 floats per batch row of output
#define DTC     0.1f

typedef __attribute__((ext_vector_type(16))) __bf16 v16bf;
typedef __attribute__((ext_vector_type(8)))  float  v8f;

union Frag16 { v16bf v; unsigned u[8]; uint4 q[2]; };

__device__ __forceinline__ unsigned short f2bf(float f) {
    unsigned x = __float_as_uint(f);
    unsigned r = 0x7FFFu + ((x >> 16) & 1u);   // round-to-nearest-even
    return (unsigned short)((x + r) >> 16);
}
__device__ __forceinline__ unsigned pack2(float lo, float hi) {
    return (unsigned)f2bf(lo) | ((unsigned)f2bf(hi) << 16);
}

// ---------------- weight fragment packing (once per launch) ----------------
// Global packed layout: block (kblk, nblk) of the K x N weight matrix is 1KB:
//   dst[((kblk*nblks + nblk)*32 + lane)*8 + j] packs bf16(W[k][col]), bf16(W[k+1][col])
//   with col = nblk*16 + lane%16, k = kblk*32 + (lane/16)*16 + 2j
// i.e. a B-operand per-lane image in this code's contiguous k order (k = 16*(lane/16) + s), same as the A frags.
__global__ void pack_w_k(const float* __restrict__ src, unsigned* __restrict__ dst,
                         int Kpad, int Ksrc, int N, int srcLd, int colOff) {
    int i = blockIdx.x * blockDim.x + threadIdx.x;
    int total = (Kpad >> 5) * (N >> 4) * 256;
    if (i >= total) return;
    int j    = i & 7;
    int lane = (i >> 3) & 31;
    int blk  = i >> 8;
    int nblks = N >> 4;
    int nblk = blk % nblks;
    int kblk = blk / nblks;
    int col = nblk * 16 + (lane & 15) + colOff;
    int k   = kblk * 32 + (lane >> 4) * 16 + 2 * j;
    float f0 = (k     < Ksrc) ? src[(size_t)k       * srcLd + col] : 0.f;
    float f1 = (k + 1 < Ksrc) ? src[(size_t)(k + 1) * srcLd + col] : 0.f;
    dst[i] = (unsigned)f2bf(f0) | ((unsigned)f2bf(f1) << 16);
}

// ---------------- s0 copy: gen[:,0,:] = states[:,0,:] ----------------
__global__ void init_s0_k(const float* __restrict__ states, float* __restrict__ gen) {
    int i = blockIdx.x * blockDim.x + threadIdx.x;       // over BATCH*NDIM
    if (i >= BATCH * NDIM) return;
    int b = i >> 7, n = i & 127;
    gen[(size_t)b * GENSTR + n] = states[(size_t)b * TSTEPS * NDIM + n];
}

__device__ __forceinline__ float zfetch(const float* __restrict__ gen,
                                        const float* __restrict__ act,
                                        int b, int t, int c) {
    if (c < NDIM)        return gen[(size_t)b * GENSTR + (size_t)t * NDIM + c];
    if (c < NDIM + NCTL) return act[(size_t)b * TSTEPS * NCTL + t * NCTL + (c - NDIM)];
    return 0.f;
}

// ---------------- fused WMMA GEMM ----------------
// MODE 0: bf16 A in, relu, bf16 out           (layer 2)
// MODE 1: A built on the fly from gen+actions, relu, bf16 out   (layer 1, fused build_z)
// MODE 2: bf16 A in, no relu, fused state update into gen       (layer 3 + update)
// WG tile 128(M) x 128(N), 8 waves: wave = (wm 0..3)*32 rows x (wn 0..1)*64 cols.
// Per K-step per wave: 4 ds_load_b128 (A frags), 8 global_load_b128 (B frags), 8 WMMA.
#define AS_STR 40    // padded LDS row stride in shorts -> conflict-free A-frag b128 reads

template<int MODE>
__global__ __launch_bounds__(256) void gemm_k(
    const unsigned short* __restrict__ Abf, int lda,
    float* __restrict__ gen, const float* __restrict__ actions,
    const unsigned* __restrict__ Wp, int nblks,
    const float* __restrict__ bias,
    unsigned short* __restrict__ outbf, int ldo,
    int ksteps, int t)
{
    __shared__ unsigned short As[128 * AS_STR];

    const int tid  = threadIdx.x;
    const int lane = tid & 31;
    const int wave = tid >> 5;
    const int wm   = wave & 3;
    const int wn   = wave >> 2;
    const int m0   = blockIdx.x * 128;
    const int n0   = blockIdx.y * 128;

    v8f acc[2][4];
    const v8f vzero = {0.f,0.f,0.f,0.f,0.f,0.f,0.f,0.f};
#pragma unroll
    for (int r = 0; r < 2; ++r)
#pragma unroll
        for (int s = 0; s < 4; ++s) acc[r][s] = vzero;

    // A staging: thread -> row (0..127), 16-col half
    const int arow  = tid >> 1;
    const int acolh = (tid & 1) * 16;
    // A fragment coords (this code's contiguous k order: lanes<16 K=0..15; lanes>=16 K=16..31)
    const int klo = (lane < 16) ? 0 : 16;
    const int am0 = (wm * 32 + (lane & 15)) * AS_STR;
    const int am1 = am0 + 16 * AS_STR;
    // B fragments straight from packed global
    const unsigned* wlane = Wp + (((size_t)(n0 >> 4) + wn * 4) * 32 + lane) * 8;

    for (int ks = 0; ks < ksteps; ++ks) {
        const int k0 = ks * 32;
        uint4 av0, av1;
        if constexpr (MODE == 1) {
            // build z = bf16([s_t, a_t, 0pad]) on the fly
            const int b = m0 + arow;
            const int cbase = k0 + acolh;
            unsigned pk[8];
            if (cbase + 16 <= NDIM) {
                const float4* gp = (const float4*)(gen + (size_t)b * GENSTR +
                                                   (size_t)t * NDIM + cbase);
#pragma unroll
                for (int q = 0; q < 4; ++q) {
                    float4 f = gp[q];
                    pk[q * 2]     = pack2(f.x, f.y);
                    pk[q * 2 + 1] = pack2(f.z, f.w);
                }
            } else {
#pragma unroll
                for (int e = 0; e < 8; ++e) {
                    int c0 = cbase + 2 * e;
                    pk[e] = pack2(zfetch(gen, actions, b, t, c0),
                                  zfetch(gen, actions, b, t, c0 + 1));
                }
            }
            av0 = *(const uint4*)&pk[0];
            av1 = *(const uint4*)&pk[4];
        } else {
            const unsigned short* ap = Abf + (size_t)(m0 + arow) * lda + k0 + acolh;
            av0 = *(const uint4*)ap;
            av1 = *(const uint4*)(ap + 8);
            if (ks + 1 < ksteps) __builtin_prefetch(ap + 32, 0, 1);  // global_prefetch_b8
        }

        __syncthreads();                       // LDS free from previous iteration
        *(uint4*)&As[arow * AS_STR + acolh]     = av0;
        *(uint4*)&As[arow * AS_STR + acolh + 8] = av1;
        __syncthreads();

        Frag16 a0f, a1f;
        a0f.q[0] = *(const uint4*)&As[am0 + klo];
        a0f.q[1] = *(const uint4*)&As[am0 + 8 + klo];
        a1f.q[0] = *(const uint4*)&As[am1 + klo];
        a1f.q[1] = *(const uint4*)&As[am1 + 8 + klo];

        const unsigned* wk = wlane + (size_t)ks * nblks * 256;
#pragma unroll
        for (int s = 0; s < 4; ++s) {
            Frag16 bfr;
            bfr.q[0] = *(const uint4*)(wk + (size_t)s * 256);
            bfr.q[1] = *(const uint4*)(wk + (size_t)s * 256 + 4);
            acc[0][s] = __builtin_amdgcn_wmma_f32_16x16x32_bf16(
                            false, a0f.v, false, bfr.v, (short)0, acc[0][s], false, false);
            acc[1][s] = __builtin_amdgcn_wmma_f32_16x16x32_bf16(
                            false, a1f.v, false, bfr.v, (short)0, acc[1][s], false, false);
        }
    }

    // epilogue: C/D layout N = lane%16, M = g + 8*(lane/16)
    const int hi = lane >> 4;
    const int nn = lane & 15;
    if constexpr (MODE != 2) {
#pragma unroll
        for (int s = 0; s < 4; ++s) {
            const int col = n0 + wn * 64 + s * 16 + nn;
            const float bv = bias[col];
#pragma unroll
            for (int r = 0; r < 2; ++r)
#pragma unroll
                for (int g = 0; g < 8; ++g) {
                    const int row = m0 + wm * 32 + r * 16 + g + 8 * hi;
                    float v = acc[r][s][g] + bv;
                    v = v > 0.f ? v : 0.f;                       // relu (MODE 0/1)
                    outbf[(size_t)row * ldo + col] = f2bf(v);
                }
        }
    } else {
        // fused: s_{t+1}[b,n] = s_t[b,n] + (P[b,2n]*a0 + P[b,2n+1]*a1)*DT
        // even lane holds col=2n, odd partner (via shfl_xor 1) holds col=2n+1
#pragma unroll
        for (int r = 0; r < 2; ++r)
#pragma unroll
            for (int g = 0; g < 8; ++g) {
                const int row = m0 + wm * 32 + r * 16 + g + 8 * hi;
                const float a0 = actions[(size_t)row * TSTEPS * NCTL + t * NCTL + 0];
                const float a1 = actions[(size_t)row * TSTEPS * NCTL + t * NCTL + 1];
#pragma unroll
                for (int s = 0; s < 4; ++s) {
                    const int col = n0 + wn * 64 + s * 16 + nn;
                    float v = acc[r][s][g] + bias[col];
                    float p = __shfl_xor(v, 1, 32);
                    if (!(lane & 1)) {
                        const int n = col >> 1;
                        float sold = gen[(size_t)row * GENSTR + (size_t)t * NDIM + n];
                        gen[(size_t)row * GENSTR + (size_t)(t + 1) * NDIM + n] =
                            sold + (v * a0 + p * a1) * DTC;
                    }
                }
            }
    }
}

// ---------------- host-side launcher ----------------
extern "C" void kernel_launch(void* const* d_in, const int* in_sizes, int n_in,
                              void* d_out, int out_size, void* d_ws, size_t ws_size,
                              hipStream_t stream) {
    const float* states  = (const float*)d_in[0];
    const float* actions = (const float*)d_in[1];
    const float* W1 = (const float*)d_in[2];
    const float* b1 = (const float*)d_in[3];
    const float* W2 = (const float*)d_in[4];
    const float* b2 = (const float*)d_in[5];
    const float* W3 = (const float*)d_in[6];
    const float* b3 = (const float*)d_in[7];
    float* gen = (float*)d_out;

    size_t off = 0;
    auto carve = [&](size_t bytes) -> void* {
        void* p = (char*)d_ws + off;
        off += (bytes + 255) & ~(size_t)255;
        return p;
    };
    unsigned*       W1p = (unsigned*)      carve((size_t)(KP1/32) * (HID/16) * 256 * 4);
    unsigned*       W2p = (unsigned*)      carve((size_t)(HID/32) * (HID/16) * 256 * 4);
    unsigned*       W3p = (unsigned*)      carve((size_t)(HID/32) * (NB /16) * 256 * 4);
    unsigned short* H1  = (unsigned short*)carve((size_t)BATCH * HID * 2);
    unsigned short* H2  = (unsigned short*)carve((size_t)BATCH * HID * 2);
    (void)ws_size; (void)in_sizes; (void)n_in; (void)out_size;

    const int TB = 256;
    const int nW1 = (KP1/32) * (HID/16) * 256;
    const int nW2 = (HID/32) * (HID/16) * 256;
    const int nW3 = (HID/32) * (NB /16) * 256;
    pack_w_k<<<(nW1 + TB - 1) / TB, TB, 0, stream>>>(W1, W1p, KP1, NDIM + NCTL, HID, HID, 0);
    pack_w_k<<<(nW2 + TB - 1) / TB, TB, 0, stream>>>(W2, W2p, HID, HID, HID, HID, 0);
    pack_w_k<<<(nW3 + TB - 1) / TB, TB, 0, stream>>>(W3, W3p, HID, HID, NB, 512, 256);
    init_s0_k<<<(BATCH * NDIM + TB - 1) / TB, TB, 0, stream>>>(states, gen);

    const dim3 gH(BATCH / 128, HID / 128);   // 32 x 8
    const dim3 g3(BATCH / 128, NB  / 128);   // 32 x 2

    for (int t = 0; t < TSTEPS; ++t) {
        // layer 1 (fused z-build): relu(z @ W1 + b1) -> H1 bf16
        gemm_k<1><<<gH, TB, 0, stream>>>(nullptr, 0, gen, actions, W1p, HID / 16,
                                         b1, H1, HID, KP1 / 32, t);
        // layer 2: relu(H1 @ W2 + b2) -> H2 bf16
        gemm_k<0><<<gH, TB, 0, stream>>>(H1, HID, nullptr, nullptr, W2p, HID / 16,
                                         b2, H2, HID, HID / 32, t);
        // layer 3 (B-part only) + fused state update into gen[:, t+1, :]
        gemm_k<2><<<g3, TB, 0, stream>>>(H2, HID, gen, actions, W3p, NB / 16,
                                         b3 + 256, nullptr, 0, HID / 32, t);
    }
}
